// IGABlock_17995912970769
// MI455X (gfx1250) — matrix operand order repaired and checked by reading, compiled, not run on hardware
//
#include <hip/hip_runtime.h>
#include <hip/hip_bf16.h>
#include <math.h>

// ---------------------------------------------------------------------------
// Types / WMMA helpers (gfx1250, wave32)
// ---------------------------------------------------------------------------
typedef __bf16 bf16t;
typedef __attribute__((ext_vector_type(16))) __bf16 v16bf;
typedef __attribute__((ext_vector_type(8)))  float  v8f;

#define NPIX 4096   // 64*64
#define CDIM 128
#define RS   20     // LDS tile row stride in uints (16B aligned, 32 bf16 + pad)
#define GRS  136    // gelu staging row stride in uints (256 bf16 + pad)

union BU { unsigned u; bf16t h[2]; };

__device__ __forceinline__ v8f wmma_bf(v16bf a, v16bf b, v8f c) {
  return __builtin_amdgcn_wmma_f32_16x16x32_bf16(false, a, false, b, (short)0, c,
                                                 false, false);
}

// A-matrix fragment (16x32 bf16, M x K), LDS tile: 16 rows x 32 halves,
// row stride rsu (uints). ISA: lanes0-15 m=0..15; VGPR i<4: K=2i+8g,
// i>=4: K=2i+8+8g  (g = lane>=16).
__device__ __forceinline__ v16bf frag_a(const unsigned* t, int rsu) {
  int lane = threadIdx.x & 31;
  int r = lane & 15, g = lane >> 4;
  const unsigned* p = t + r * rsu;
  v16bf f;
#pragma unroll
  for (int i = 0; i < 8; ++i) {
    int ku = (i < 4) ? (i + 4 * g) : (i + 4 + 4 * g);
    BU u; u.u = p[ku];
    f[2 * i] = u.h[0];
    f[2 * i + 1] = u.h[1];
  }
  return f;
}

// B-matrix fragment (32x16 bf16, K x N), LDS tile stored as [n(16)][k(32)].
// Same K order as frag_a: VGPR i<4: K=2i+8g, i>=4: K=2i+8+8g (g = lane>=16).
__device__ __forceinline__ v16bf frag_b(const unsigned* t, int rsu) {
  int lane = threadIdx.x & 31;
  int r = lane & 15, g = lane >> 4;
  const unsigned* p = t + r * rsu;
  v16bf f;
#pragma unroll
  for (int i = 0; i < 8; ++i) {
    BU u; u.u = p[(i < 4) ? (i + 4 * g) : (i + 4 + 4 * g)];
    f[2 * i] = u.h[0];
    f[2 * i + 1] = u.h[1];
  }
  return f;
}

__device__ __forceinline__ v8f vzero8() {
  v8f z = {0.f, 0.f, 0.f, 0.f, 0.f, 0.f, 0.f, 0.f};
  return z;
}

__device__ __forceinline__ unsigned pack2bf(float a, float b) {
  BU u; u.h[0] = (bf16t)a; u.h[1] = (bf16t)b; return u.u;
}

// ---------------------------------------------------------------------------
// Prep: x (B,C,N) f32 -> xT (B,N,C) f32 + bf16   (tiled transpose)
// ---------------------------------------------------------------------------
__global__ void __launch_bounds__(256)
k_transpose(const float* __restrict__ x, float* __restrict__ xT,
            bf16t* __restrict__ xTb) {
  __shared__ float tile[32][33];
  int b = blockIdx.z, c0 = blockIdx.y * 32, n0 = blockIdx.x * 32;
  int tx = threadIdx.x, ty = threadIdx.y;  // 32 x 8
#pragma unroll
  for (int i = 0; i < 4; ++i)
    tile[ty + 8 * i][tx] = x[(b * CDIM + c0 + ty + 8 * i) * NPIX + n0 + tx];
  __syncthreads();
#pragma unroll
  for (int i = 0; i < 4; ++i) {
    int n = n0 + ty + 8 * i;
    float v = tile[tx][ty + 8 * i];
    xT[(b * NPIX + n) * CDIM + c0 + tx] = v;
    xTb[(b * NPIX + n) * CDIM + c0 + tx] = (bf16t)v;
  }
}

// Generic f32 -> bf16 convert
__global__ void k_cvt(const float* __restrict__ s, bf16t* __restrict__ d, int n) {
  int i = blockIdx.x * blockDim.x + threadIdx.x;
  if (i < n) d[i] = (bf16t)s[i];
}

// f1_w (oc,ic,3,3) f32 -> f1p (tap, oc, ic) bf16
__global__ void k_packf1(const float* __restrict__ w, bf16t* __restrict__ o) {
  int i = blockIdx.x * blockDim.x + threadIdx.x;
  if (i >= 9 * 256 * 128) return;
  int tap = i / (256 * 128);
  int rem = i - tap * (256 * 128);
  int oc = rem >> 7, ic = rem & 127;
  o[i] = (bf16t)w[(oc * 128 + ic) * 9 + tap];
}

// ---------------------------------------------------------------------------
// QKV: Y = W(384x128) @ X(128xN) + b   -> q (scaled), k, v bf16
// q,k: (B,H,N,32)  v: (B,H,32,N)
// Grid: (N/64, 3, B) block 256 (8 waves). Wave w owns oc-subtile w of 128-block.
// ---------------------------------------------------------------------------
__global__ void __launch_bounds__(256)
k_qkv(const bf16t* __restrict__ xTb, const bf16t* __restrict__ wb,
      const float* __restrict__ bias, bf16t* __restrict__ qT,
      bf16t* __restrict__ kT, bf16t* __restrict__ vT) {
  __shared__ unsigned Alds[128 * RS];
  __shared__ unsigned Blds[64 * RS];
  int tid = threadIdx.x, lane = tid & 31, w = tid >> 5;
  int n0 = blockIdx.x * 64, gy = blockIdx.y, b = blockIdx.z;
  const uint4* wu = (const uint4*)wb;   // row = 16 uints = 4 quads
  const uint4* xu = (const uint4*)xTb;
  v8f acc[4];
#pragma unroll
  for (int j = 0; j < 4; ++j) acc[j] = vzero8();

  for (int kc = 0; kc < 4; ++kc) {
    __syncthreads();
#pragma unroll
    for (int it = 0; it < 2; ++it) {            // 128 rows * 4 quads
      int idx = tid + it * 256;
      int row = idx >> 2, q = idx & 3;
      *(uint4*)(Alds + row * RS + q * 4) = wu[(gy * 128 + row) * 16 + kc * 4 + q];
    }
    {                                           // 64 rows * 4 quads
      int row = tid >> 2, q = tid & 3;
      *(uint4*)(Blds + row * RS + q * 4) =
          xu[(b * NPIX + n0 + row) * 16 + kc * 4 + q];
    }
    __syncthreads();
    v16bf af = frag_a(Alds + (w * 16) * RS, RS);
#pragma unroll
    for (int j = 0; j < 4; ++j) {
      v16bf bf = frag_b(Blds + (j * 16) * RS, RS);
      acc[j] = wmma_bf(af, bf, acc[j]);
    }
  }
  const float qscale = 0.17677669529663687f;  // 1/sqrt(32)
  int nl = lane & 15, g = lane >> 4;
#pragma unroll
  for (int j = 0; j < 4; ++j) {
    int np = n0 + j * 16 + nl;
#pragma unroll
    for (int r = 0; r < 8; ++r) {
      int m = w * 16 + 8 * g + r;
      float v = acc[j][r] + bias[gy * 128 + m];
      int h = m >> 5, d = m & 31;
      int bh = b * 4 + h;
      if (gy == 0)      qT[(bh * NPIX + np) * 32 + d] = (bf16t)(v * qscale);
      else if (gy == 1) kT[(bh * NPIX + np) * 32 + d] = (bf16t)v;
      else              vT[(bh * 32 + d) * NPIX + np] = (bf16t)v;
    }
  }
}

// ---------------------------------------------------------------------------
// Flash attention per (b,h). Grid (N/64, H, B), block 128 (4 waves,
// wave = 16 queries). Writes x1T = x + attn_out, bf16 (B,N,C).
// ---------------------------------------------------------------------------
__global__ void __launch_bounds__(128)
k_attn(const bf16t* __restrict__ qT, const bf16t* __restrict__ kT,
       const bf16t* __restrict__ vT, const float* __restrict__ xT,
       bf16t* __restrict__ x1Tb) {
  __shared__ unsigned Qlds[64 * RS];
  __shared__ unsigned Klds[32 * RS];
  __shared__ unsigned Vlds[32 * RS];
  __shared__ unsigned Plds[4 * 16 * 17];
  int tid = threadIdx.x, lane = tid & 31, w = tid >> 5;
  int q0 = blockIdx.x * 64, h = blockIdx.y, b = blockIdx.z;
  int bh = b * 4 + h;
  const uint4* qu = (const uint4*)qT;
  const uint4* ku = (const uint4*)kT;
  const uint4* vu = (const uint4*)vT;

#pragma unroll
  for (int it = 0; it < 2; ++it) {              // 64 rows * 4 quads
    int idx = tid + it * 128;
    int row = idx >> 2, q = idx & 3;
    *(uint4*)(Qlds + row * RS + q * 4) = qu[(bh * NPIX + q0 + row) * 4 + q];
  }
  __syncthreads();
  v16bf qf = frag_a(Qlds + (w * 16) * RS, RS);

  v8f o0 = vzero8(), o1 = vzero8();
  float M[8], L[8];
#pragma unroll
  for (int r = 0; r < 8; ++r) { M[r] = -1e30f; L[r] = 0.f; }

  int nl = lane & 15, g = lane >> 4;
  bf16t* Pw = (bf16t*)(Plds + w * 16 * 17);

  for (int kc = 0; kc < 128; ++kc) {
    {                                           // 32 rows * 4 quads each
      int row = tid >> 2, q = tid & 3;
      *(uint4*)(Klds + row * RS + q * 4) =
          ku[(bh * NPIX + kc * 32 + row) * 4 + q];
      *(uint4*)(Vlds + row * RS + q * 4) =
          vu[(bh * 32 + row) * 512 + kc * 4 + q];
    }
    __syncthreads();
    v16bf bk0 = frag_b(Klds, RS);
    v16bf bk1 = frag_b(Klds + 16 * RS, RS);
    v16bf bv0 = frag_b(Vlds, RS);
    v16bf bv1 = frag_b(Vlds + 16 * RS, RS);
    v8f s0 = vzero8(), s1 = vzero8();
    s0 = wmma_bf(qf, bk0, s0);   // S[q, key0..15]   (q pre-scaled)
    s1 = wmma_bf(qf, bk1, s1);   // S[q, key16..31]
#pragma unroll
    for (int r = 0; r < 8; ++r) {
      float mx = fmaxf(s0[r], s1[r]);
      mx = fmaxf(mx, __shfl_xor(mx, 1, 32));
      mx = fmaxf(mx, __shfl_xor(mx, 2, 32));
      mx = fmaxf(mx, __shfl_xor(mx, 4, 32));
      mx = fmaxf(mx, __shfl_xor(mx, 8, 32));
      float mn = fmaxf(M[r], mx);
      float alpha = __expf(M[r] - mn);
      float p0 = __expf(s0[r] - mn);
      float p1 = __expf(s1[r] - mn);
      float rs = p0 + p1;
      rs += __shfl_xor(rs, 1, 32);
      rs += __shfl_xor(rs, 2, 32);
      rs += __shfl_xor(rs, 4, 32);
      rs += __shfl_xor(rs, 8, 32);
      L[r] = L[r] * alpha + rs;
      M[r] = mn;
      o0[r] *= alpha;
      o1[r] *= alpha;
      Pw[(8 * g + r) * 34 + nl] = (bf16t)p0;       // P[q][key] -> A layout
      Pw[(8 * g + r) * 34 + 16 + nl] = (bf16t)p1;
    }
    __syncthreads();
    v16bf pf = frag_a(Plds + w * 16 * 17, 17);
    o0 = wmma_bf(pf, bv0, o0);   // O[q, d0..15]
    o1 = wmma_bf(pf, bv1, o1);   // O[q, d16..31]
  }
  // C layout of O: n = d, m = q
  int d = nl;
#pragma unroll
  for (int r = 0; r < 8; ++r) {
    int q = q0 + w * 16 + 8 * g + r;
    int base = (b * NPIX + q) * CDIM + h * 32;
    float inv = 1.0f / L[r];
    x1Tb[base + d] = (bf16t)(xT[base + d] + o0[r] * inv);
    x1Tb[base + 16 + d] = (bf16t)(xT[base + 16 + d] + o1[r] * inv);
  }
}

// ---------------------------------------------------------------------------
// Channel shuffle: x2[:, g*32+j] = wg[g] @ x1[:, g*32: g*32+32] + cs_b
// Grid (N/64, G=4, B), block 128. Output x2T f32 (B,N,C).
// ---------------------------------------------------------------------------
__global__ void __launch_bounds__(128)
k_shuffle(const bf16t* __restrict__ cswb, const bf16t* __restrict__ x1Tb,
          const float* __restrict__ csb, float* __restrict__ x2T) {
  __shared__ unsigned Alds[32 * RS];
  __shared__ unsigned Blds[64 * RS];
  int tid = threadIdx.x, lane = tid & 31, w = tid >> 5;
  int n0 = blockIdx.x * 64, g = blockIdx.y, b = blockIdx.z;
  const uint4* au = (const uint4*)cswb;
  const uint4* xu = (const uint4*)x1Tb;
  {                                             // 32 rows * 4 quads
    int row = tid >> 2, q = tid & 3;
    *(uint4*)(Alds + row * RS + q * 4) = au[(g * 32 + row) * 4 + q];
  }
#pragma unroll
  for (int it = 0; it < 2; ++it) {              // 64 rows * 4 quads
    int idx = tid + it * 128;
    int row = idx >> 2, q = idx & 3;
    *(uint4*)(Blds + row * RS + q * 4) =
        xu[(b * NPIX + n0 + row) * 16 + g * 4 + q];
  }
  __syncthreads();
  v16bf bf = frag_b(Blds + (w * 16) * RS, RS);
  v8f a0 = vzero8(), a1 = vzero8();
  a0 = wmma_bf(frag_a(Alds, RS), bf, a0);
  a1 = wmma_bf(frag_a(Alds + 16 * RS, RS), bf, a1);
  int nl = lane & 15, gg = lane >> 4;
  int np = n0 + w * 16 + nl;
  float* dst = x2T + (b * NPIX + np) * CDIM + g * 32;
#pragma unroll
  for (int r = 0; r < 8; ++r) {
    int m0 = 8 * gg + r;
    dst[m0] = a0[r] + csb[g * 32 + m0];
    dst[16 + m0] = a1[r] + csb[g * 32 + 16 + m0];
  }
}

// ---------------------------------------------------------------------------
// LayerNorm over C=128 per pixel. 4 threads/pixel, shfl reduction.
// ---------------------------------------------------------------------------
__global__ void __launch_bounds__(256)
k_ln(const float* __restrict__ x2T, const float* __restrict__ lnw,
     const float* __restrict__ lnb, float* __restrict__ xlnT,
     bf16t* __restrict__ xlnTb) {
  int t = blockIdx.x * 256 + threadIdx.x;
  int pix = t >> 2, q = t & 3;
  const float4* src = (const float4*)(x2T + pix * CDIM + q * 32);
  float4 v[8];
  float s = 0.f, ss = 0.f;
#pragma unroll
  for (int i = 0; i < 8; ++i) {
    v[i] = src[i];
    s += v[i].x + v[i].y + v[i].z + v[i].w;
    ss += v[i].x * v[i].x + v[i].y * v[i].y + v[i].z * v[i].z + v[i].w * v[i].w;
  }
  s += __shfl_xor(s, 1, 32);  s += __shfl_xor(s, 2, 32);
  ss += __shfl_xor(ss, 1, 32); ss += __shfl_xor(ss, 2, 32);
  float mu = s * (1.f / 128.f);
  float var = ss * (1.f / 128.f) - mu * mu;
  float rstd = rsqrtf(var + 1e-5f);
  float* dst = xlnT + pix * CDIM + q * 32;
  bf16t* dstb = xlnTb + pix * CDIM + q * 32;
#pragma unroll
  for (int i = 0; i < 8; ++i) {
    float e[4] = {v[i].x, v[i].y, v[i].z, v[i].w};
#pragma unroll
    for (int j = 0; j < 4; ++j) {
      int c = q * 32 + i * 4 + j;
      float o = (e[j] - mu) * rstd * lnw[c] + lnb[c];
      dst[i * 4 + j] = o;
      dstb[i * 4 + j] = (bf16t)o;
    }
  }
}

// ---------------------------------------------------------------------------
// Fused 3x3 conv (implicit GEMM, 128->256) + GELU + 1x1 (256->128) + residual.
// Grid (N/32, B), block 256 (8 waves). Wave w owns oc {32w..32w+31} (conv)
// and out-ch {16w..16w+15} (f2); 2 pixel-subtiles of 16.
// ---------------------------------------------------------------------------
__global__ void __launch_bounds__(256)
k_convmlp(const bf16t* __restrict__ f1p, const float* __restrict__ f1b,
          const bf16t* __restrict__ f2wb, const float* __restrict__ f2b,
          const bf16t* __restrict__ xlnTb, const float* __restrict__ xlnT,
          float* __restrict__ out) {
  __shared__ unsigned A1[256 * RS];   // conv weight tiles (all 256 oc)
  __shared__ unsigned B1[32 * RS];    // input tiles (2 x 16 px)
  __shared__ unsigned GL[32 * GRS];   // gelu activations [px32][oc]
  __shared__ unsigned A2[128 * RS];   // f2 weight tiles
  int tid = threadIdx.x, lane = tid & 31, w = tid >> 5;
  int n0 = blockIdx.x * 32, b = blockIdx.y;
  int y = n0 >> 6, x0 = n0 & 63;
  const uint4* f1u = (const uint4*)f1p;
  const uint4* xu = (const uint4*)xlnTb;
  int nl = lane & 15, g = lane >> 4;

  v8f acc[2][2];
#pragma unroll
  for (int s = 0; s < 2; ++s)
#pragma unroll
    for (int j = 0; j < 2; ++j) acc[s][j] = vzero8();

  for (int tap = 0; tap < 9; ++tap) {
    int dy = tap / 3 - 1, dx = tap % 3 - 1;
    int yy = y + dy;
    for (int kc = 0; kc < 4; ++kc) {
      __syncthreads();
#pragma unroll
      for (int it = 0; it < 4; ++it) {          // 256 rows * 4 quads
        int idx = tid + it * 256;
        int row = idx >> 2, q = idx & 3;
        *(uint4*)(A1 + row * RS + q * 4) =
            f1u[(tap * 256 + row) * 16 + kc * 4 + q];
      }
      if (tid < 128) {                          // 32 rows * 4 quads
        int row = tid >> 2, q = tid & 3;
        int xx = x0 + row + dx;
        uint4 val = {0u, 0u, 0u, 0u};
        if (yy >= 0 && yy < 64 && xx >= 0 && xx < 64)
          val = xu[(b * NPIX + yy * 64 + xx) * 16 + kc * 4 + q];
        *(uint4*)(B1 + row * RS + q * 4) = val;
      }
      __syncthreads();
      v16bf bf0 = frag_b(B1, RS);
      v16bf bf1 = frag_b(B1 + 16 * RS, RS);
#pragma unroll
      for (int s = 0; s < 2; ++s) {
        v16bf af = frag_a(A1 + ((w * 2 + s) * 16) * RS, RS);
        acc[s][0] = wmma_bf(af, bf0, acc[s][0]);
        acc[s][1] = wmma_bf(af, bf1, acc[s][1]);
      }
    }
  }
  // bias + GELU(exact) -> stage to LDS in [px][oc] (B-tile) layout, b128 stores
#pragma unroll
  for (int s = 0; s < 2; ++s)
#pragma unroll
    for (int j = 0; j < 2; ++j) {
      int oc0 = (w * 2 + s) * 16 + 8 * g;
      float ge[8];
#pragma unroll
      for (int r = 0; r < 8; ++r) {
        float hv = acc[s][j][r] + f1b[oc0 + r];
        ge[r] = 0.5f * hv * (1.0f + erff(hv * 0.70710678118654752f));
      }
      uint4 pk;
      pk.x = pack2bf(ge[0], ge[1]);
      pk.y = pack2bf(ge[2], ge[3]);
      pk.z = pack2bf(ge[4], ge[5]);
      pk.w = pack2bf(ge[6], ge[7]);
      *(uint4*)(GL + (j * 16 + nl) * GRS + (oc0 >> 1)) = pk;
    }
  // f2 projection: 128 x 256
  const uint4* f2u = (const uint4*)f2wb;
  v8f a2[2];
  a2[0] = vzero8(); a2[1] = vzero8();
  for (int kc = 0; kc < 8; ++kc) {
    __syncthreads();
#pragma unroll
    for (int it = 0; it < 2; ++it) {            // 128 rows * 4 quads
      int idx = tid + it * 256;
      int row = idx >> 2, q = idx & 3;
      *(uint4*)(A2 + row * RS + q * 4) = f2u[row * 32 + kc * 4 + q];
    }
    __syncthreads();
    v16bf af = frag_a(A2 + (w * 16) * RS, RS);
#pragma unroll
    for (int j = 0; j < 2; ++j) {
      v16bf bf = frag_b(GL + (j * 16) * GRS + kc * 16, GRS);
      a2[j] = wmma_bf(af, bf, a2[j]);
    }
  }
  // bias + LN residual, store (B,C,H,W)
#pragma unroll
  for (int j = 0; j < 2; ++j) {
    int np = n0 + j * 16 + nl;
#pragma unroll
    for (int r = 0; r < 8; ++r) {
      int o = w * 16 + 8 * g + r;
      float v = a2[j][r] + f2b[o] + xlnT[(b * NPIX + np) * CDIM + o];
      out[(b * CDIM + o) * NPIX + np] = v;
    }
  }
}

// ---------------------------------------------------------------------------
// Host launcher
// ---------------------------------------------------------------------------
extern "C" void kernel_launch(void* const* d_in, const int* in_sizes, int n_in,
                              void* d_out, int out_size, void* d_ws,
                              size_t ws_size, hipStream_t stream) {
  const float* x      = (const float*)d_in[0];
  const float* qkv_w  = (const float*)d_in[1];
  const float* qkv_b  = (const float*)d_in[2];
  const float* cs_w   = (const float*)d_in[3];
  const float* cs_b   = (const float*)d_in[4];
  const float* ln_w   = (const float*)d_in[5];
  const float* ln_b   = (const float*)d_in[6];
  const float* f1_w   = (const float*)d_in[7];
  const float* f1_b   = (const float*)d_in[8];
  const float* f2_w   = (const float*)d_in[9];
  const float* f2_b   = (const float*)d_in[10];

  char* ws = (char*)d_ws;
  size_t off = 0;
  auto alloc = [&](size_t bytes) -> void* {
    void* p = ws + off;
    off = (off + bytes + 255) & ~(size_t)255;
    return p;
  };
  const int B = 2;
  float* xT    = (float*)alloc((size_t)B * NPIX * CDIM * 4);
  bf16t* xTb   = (bf16t*)alloc((size_t)B * NPIX * CDIM * 2);
  bf16t* qT    = (bf16t*)alloc((size_t)B * 4 * NPIX * 32 * 2);
  bf16t* kT    = (bf16t*)alloc((size_t)B * 4 * NPIX * 32 * 2);
  bf16t* vT    = (bf16t*)alloc((size_t)B * 4 * 32 * NPIX * 2);
  bf16t* x1Tb  = (bf16t*)alloc((size_t)B * NPIX * CDIM * 2);
  float* x2T   = (float*)alloc((size_t)B * NPIX * CDIM * 4);
  float* xlnT  = (float*)alloc((size_t)B * NPIX * CDIM * 4);
  bf16t* xlnTb = (bf16t*)alloc((size_t)B * NPIX * CDIM * 2);
  bf16t* qkvwb = (bf16t*)alloc((size_t)384 * 128 * 2);
  bf16t* cswb  = (bf16t*)alloc((size_t)128 * 32 * 2);
  bf16t* f1p   = (bf16t*)alloc((size_t)9 * 256 * 128 * 2);
  bf16t* f2wb  = (bf16t*)alloc((size_t)128 * 256 * 2);
  (void)ws_size; (void)n_in; (void)in_sizes; (void)out_size;

  k_transpose<<<dim3(128, 4, 2), dim3(32, 8), 0, stream>>>(x, xT, xTb);
  k_cvt<<<(49152 + 255) / 256, 256, 0, stream>>>(qkv_w, qkvwb, 49152);
  k_cvt<<<(4096 + 255) / 256, 256, 0, stream>>>(cs_w, cswb, 4096);
  k_cvt<<<(32768 + 255) / 256, 256, 0, stream>>>(f2_w, f2wb, 32768);
  k_packf1<<<(294912 + 255) / 256, 256, 0, stream>>>(f1_w, f1p);

  k_qkv<<<dim3(64, 3, 2), 256, 0, stream>>>(xTb, qkvwb, qkv_b, qT, kT, vT);
  k_attn<<<dim3(64, 4, 2), 128, 0, stream>>>(qT, kT, vT, xT, x1Tb);
  k_shuffle<<<dim3(64, 4, 2), 128, 0, stream>>>(cswb, x1Tb, cs_b, x2T);
  k_ln<<<128, 256, 0, stream>>>(x2T, ln_w, ln_b, xlnT, xlnTb);
  k_convmlp<<<dim3(128, 2), 256, 0, stream>>>(f1p, f1_b, f2wb, f2_b, xlnTb,
                                              xlnT, (float*)d_out);
}
